// LocalTransformerBlock2d_46110768890400
// MI455X (gfx1250) — matrix operand order repaired and checked by reading, not compiled, not run on hardware
//
#include <hip/hip_runtime.h>
#include <math.h>

// ---------------------------------------------------------------------------
// LocalTransformerBlock2d for MI455X (gfx1250, wave32, WMMA bf16)
// GEMMs: v_wmma_f32_16x16x32_bf16, fp32 accumulate, bf16 activations in
// pixel-major (B,P,C) layout. B tiles staged through LDS (async when the
// toolchain exposes global_load_async_to_lds), double buffered.
// Workspace layout (~640.4 MB):
//   region A @ 0        : 128 MB bf16 (B,P,128)   xln / attn_out / xln2
//   region B @ 128 MB   : 512 MB bf16 (B,P,<=512) qkv / mlp hidden
//   region C @ 640 MB   : 384 KB bf16 pre-converted weights
// ---------------------------------------------------------------------------

typedef __attribute__((ext_vector_type(16))) __bf16 v16bf;
typedef __attribute__((ext_vector_type(8)))  __bf16 v8bf;
typedef __attribute__((ext_vector_type(8)))  float  v8f;
typedef __attribute__((ext_vector_type(4)))  int    v4i;

#define HW  65536          // 256*256
#define IMW 256
#define NB  8
#define C_IN 128

#if __has_builtin(__builtin_amdgcn_global_load_async_to_lds_b128)
#define HAVE_ASYNC_LDS 1
#else
#define HAVE_ASYNC_LDS 0
#endif

// 16-byte global -> LDS copy (async on CDNA5 when available)
__device__ __forceinline__ void cp16_g2l(const __bf16* g, __bf16* l) {
#if HAVE_ASYNC_LDS
    __builtin_amdgcn_global_load_async_to_lds_b128(
        (__attribute__((address_space(1))) v4i*)(g),
        (__attribute__((address_space(3))) v4i*)(l), 0, 0);
#else
    *(v8bf*)l = *(const v8bf*)g;
#endif
}

template <int N>
__device__ __forceinline__ void async_wait() {
#if HAVE_ASYNC_LDS
#if __has_builtin(__builtin_amdgcn_s_wait_asynccnt)
    __builtin_amdgcn_s_wait_asynccnt(N);
#else
    asm volatile("s_wait_asynccnt %0" ::"n"(N));
#endif
#endif
}

__device__ __forceinline__ float gelu_tanh(float v) {
    return 0.5f * v * (1.0f + tanhf(0.7978845608028654f * (v + 0.044715f * v * v * v)));
}

// -------------------- weight fp32 -> bf16 pre-conversion -------------------
__global__ __launch_bounds__(256) void cvt_w_kernel(const float* __restrict__ src,
                                                    __bf16* __restrict__ dst, int n) {
    int i = blockIdx.x * 256 + threadIdx.x;
    if (i < n) dst[i] = (__bf16)src[i];
}

// -------------------- channel LayerNorm: (B,C,HW) f32 -> (B,P,C) bf16 ------
__global__ __launch_bounds__(256) void ln_kernel(const float* __restrict__ X,
                                                 const float* __restrict__ g,
                                                 const float* __restrict__ be,
                                                 __bf16* __restrict__ Y) {
    size_t idx = (size_t)blockIdx.x * 256 + threadIdx.x;   // b*HW + p
    size_t b = idx >> 16;
    size_t p = idx & 65535u;
    const float* xp = X + b * (size_t)C_IN * HW + p;       // channel stride HW
    float s = 0.f, s2 = 0.f;
    for (int c = 0; c < C_IN; ++c) {
        float v = xp[(size_t)c * HW];
        s += v; s2 += v * v;
    }
    float mu   = s * (1.0f / 128.0f);
    float var  = s2 * (1.0f / 128.0f) - mu * mu;
    float rstd = rsqrtf(var + 1e-5f);
    __bf16* yp = Y + idx * C_IN;
    for (int c0 = 0; c0 < C_IN; c0 += 8) {
        v8bf pk;
#pragma unroll
        for (int j = 0; j < 8; ++j) {
            float v = xp[(size_t)(c0 + j) * HW];
            v = (v - mu) * rstd * g[c0 + j] + be[c0 + j];
            pk[j] = (__bf16)v;
        }
        *(v8bf*)(yp + c0) = pk;
    }
}

// -------------------- bf16 WMMA GEMM: Out[O,P] = W[O,Cin] * X[P,Cin]^T -----
// Block = 8 waves: wave w owns O-tile (blockIdx.x*8+w), all share one
// 64-pixel P tile whose 32xK slab is staged in LDS (double buffered).
// MODE 0: bf16 out (B,P,O) + bias        (QKV)
// MODE 1: bf16 out (B,P,O) + bias + GELU (MLP1)
// MODE 2: f32  out (B,O,HW) = residual + acc + bias (proj / MLP2)
#define BPITCH 40   // LDS row pitch in bf16 (80B: 16B-aligned, bank-spread)
template <int MODE>
__global__ __launch_bounds__(256) void gemm_wmma(const __bf16* __restrict__ Wb,
                                                 const __bf16* __restrict__ Xb,
                                                 const float*  __restrict__ bias,
                                                 __bf16*       __restrict__ Obf,
                                                 const float*  __restrict__ Res,
                                                 float*        __restrict__ Of,
                                                 int Cin, int Ot) {
    __shared__ __bf16 sB[2][64 * BPITCH];

    const int tid  = threadIdx.x;
    const int lane = tid & 31;
    const int wave = tid >> 5;
    const int b    = blockIdx.z;
    const int p0   = blockIdx.y * 64;
    const int ot   = blockIdx.x * 8 + wave;   // this wave's 16-wide O tile
    const int n    = lane & 15;               // column (pixel) owned by lane
    const int hi   = lane >> 4;               // K-half / M-half selector
    const int o_lo = ot * 16 + hi * 8;

    // cooperative staging: thread -> (pixel row, 16B k-chunk)
    const int srow   = tid >> 2;              // 0..63
    const int schunk = tid & 3;               // 0..3 (8 bf16 each)
    const __bf16* gsrc = Xb + ((size_t)b * HW + p0 + srow) * (size_t)Cin + schunk * 8;
    __bf16* ldst[2] = { &sB[0][srow * BPITCH + schunk * 8],
                        &sB[1][srow * BPITCH + schunk * 8] };

    float bias8[8];
#pragma unroll
    for (int r = 0; r < 8; ++r) bias8[r] = bias[o_lo + r];

    v8f acc[4];
#pragma unroll
    for (int t = 0; t < 4; ++t)
#pragma unroll
        for (int r = 0; r < 8; ++r) acc[t][r] = 0.f;

    // A fragment: bf16 weight row M = ot*16 + n, K offset hi*8 (+16)
    const __bf16* wrow = Wb + (size_t)(ot * 16 + n) * Cin + hi * 8;

    const int nk = Cin >> 5;
    cp16_g2l(gsrc, ldst[0]);                       // prime k-step 0

    for (int kk = 0; kk < nk; ++kk) {
        if (kk + 1 < nk) {                         // prefetch next slab
            cp16_g2l(gsrc + (size_t)(kk + 1) * 32, ldst[(kk + 1) & 1]);
            async_wait<1>();
        } else {
            async_wait<0>();
        }
        __syncthreads();                           // slab kk visible to all

        v8bf a0 = *(const v8bf*)(wrow + kk * 32);
        v8bf a1 = *(const v8bf*)(wrow + kk * 32 + 16);
        v16bf A = __builtin_shufflevector(a0, a1,
                      0,1,2,3,4,5,6,7,8,9,10,11,12,13,14,15);
        const __bf16* sb = sB[kk & 1];
#pragma unroll
        for (int t = 0; t < 4; ++t) {
            const __bf16* bp = sb + (t * 16 + n) * BPITCH + hi * 8;
            v8bf blo = *(const v8bf*)bp;
            v8bf bhi = *(const v8bf*)(bp + 16);
            v16bf B = __builtin_shufflevector(blo, bhi,
                          0,1,2,3,4,5,6,7,8,9,10,11,12,13,14,15);
            acc[t] = __builtin_amdgcn_wmma_f32_16x16x32_bf16(
                         false, A, false, B, (short)0, acc[t], false, false);
        }
        __syncthreads();                           // done reading slab kk
    }

    if (MODE == 2) {
#pragma unroll
        for (int t = 0; t < 4; ++t) {
            const int p = p0 + t * 16 + n;
#pragma unroll
            for (int r = 0; r < 8; ++r) {
                size_t addr = ((size_t)b * Ot + o_lo + r) * (size_t)HW + p;
                Of[addr] = Res[addr] + acc[t][r] + bias8[r];
            }
        }
    } else {
#pragma unroll
        for (int t = 0; t < 4; ++t) {
            size_t prow = (size_t)b * HW + p0 + t * 16 + n;
            v8bf pk;
#pragma unroll
            for (int r = 0; r < 8; ++r) {
                float v = acc[t][r] + bias8[r];
                if (MODE == 1) v = gelu_tanh(v);
                pk[r] = (__bf16)v;
            }
            *(v8bf*)(Obf + prow * Ot + o_lo) = pk;
        }
    }
}

// -------------------- 7x7 local attention, online softmax ------------------
// QKV: (B, P, 384) bf16 [q|k|v], out: (B, P, 128) bf16 (channel = head*32+j)
__global__ __launch_bounds__(256) void attn_kernel(const __bf16* __restrict__ QKV,
                                                   __bf16* __restrict__ O) {
    size_t idx = (size_t)blockIdx.x * 256 + threadIdx.x;   // b*4*HW + head*HW + p
    const int b    = (int)(idx >> 18);
    const int head = (int)((idx >> 16) & 3);
    const int p    = (int)(idx & 65535u);
    const int y = p >> 8, x = p & 255;

    const __bf16* base = QKV + (size_t)b * HW * 384;
    const __bf16* qp = base + (size_t)p * 384 + head * 32;
    float q[32];
#pragma unroll
    for (int j = 0; j < 32; ++j) q[j] = (float)qp[j] * 0.17677669529663687f;

    float m = -1e30f, s = 0.f;
    float oacc[32];
#pragma unroll
    for (int j = 0; j < 32; ++j) oacc[j] = 0.f;

    for (int oi = 0; oi < 7; ++oi) {
        const int ny = y + oi - 3;
        const bool vy = (ny >= 0) && (ny < IMW);
        for (int oj = 0; oj < 7; ++oj) {
            const int nx = x + oj - 3;
            const bool valid = vy && (nx >= 0) && (nx < IMW);
            float sc = 0.f;
            const __bf16* kp = nullptr;
            if (valid) {
                kp = base + ((size_t)(ny * IMW + nx)) * 384 + 128 + head * 32;
#pragma unroll
                for (int j = 0; j < 32; ++j) sc += q[j] * (float)kp[j];
            }
            // padded taps: score 0 (q.0), v = 0, but still inside the softmax
            float wgt;
            if (sc > m) {
                float f = __expf(m - sc);
                s = s * f + 1.f;
#pragma unroll
                for (int j = 0; j < 32; ++j) oacc[j] *= f;
                m = sc; wgt = 1.f;
            } else {
                wgt = __expf(sc - m);
                s += wgt;
            }
            if (valid) {
                const __bf16* vp = kp + 128;
#pragma unroll
                for (int j = 0; j < 32; ++j) oacc[j] += wgt * (float)vp[j];
            }
        }
    }
    const float inv = 1.f / s;
    __bf16* op = O + ((size_t)b * HW + p) * 128 + head * 32;
#pragma unroll
    for (int j = 0; j < 32; ++j) op[j] = (__bf16)(oacc[j] * inv);
}

// ---------------------------------------------------------------------------
extern "C" void kernel_launch(void* const* d_in, const int* in_sizes, int n_in,
                              void* d_out, int out_size, void* d_ws, size_t ws_size,
                              hipStream_t stream) {
    (void)in_sizes; (void)n_in; (void)out_size; (void)ws_size;
    const float* x      = (const float*)d_in[0];
    const float* n1_w   = (const float*)d_in[1];
    const float* n1_b   = (const float*)d_in[2];
    const float* qkv_w  = (const float*)d_in[3];
    const float* qkv_b  = (const float*)d_in[4];
    const float* proj_w = (const float*)d_in[5];
    const float* proj_b = (const float*)d_in[6];
    const float* n2_w   = (const float*)d_in[7];
    const float* n2_b   = (const float*)d_in[8];
    const float* mlp_w1 = (const float*)d_in[9];
    const float* mlp_b1 = (const float*)d_in[10];
    const float* mlp_w2 = (const float*)d_in[11];
    const float* mlp_b2 = (const float*)d_in[12];
    float* out = (float*)d_out;

    __bf16* regA = (__bf16*)d_ws;                                   // 128 MB
    __bf16* regB = (__bf16*)((char*)d_ws + (size_t)134217728);      // 512 MB
    __bf16* regC = (__bf16*)((char*)d_ws + (size_t)671088640);      // weights

    // pre-converted bf16 weights
    __bf16* qkvWb  = regC;              // 384*128 = 49152
    __bf16* projWb = regC + 49152;      // 128*128 = 16384
    __bf16* mlp1Wb = regC + 65536;      // 512*128 = 65536
    __bf16* mlp2Wb = regC + 131072;     // 128*512 = 65536

    dim3 blk(256);

    cvt_w_kernel<<<192, blk, 0, stream>>>(qkv_w,  qkvWb,  49152);
    cvt_w_kernel<<<64,  blk, 0, stream>>>(proj_w, projWb, 16384);
    cvt_w_kernel<<<256, blk, 0, stream>>>(mlp_w1, mlp1Wb, 65536);
    cvt_w_kernel<<<256, blk, 0, stream>>>(mlp_w2, mlp2Wb, 65536);

    // x -> LN1 -> xln (A)
    ln_kernel<<<2048, blk, 0, stream>>>(x, n1_w, n1_b, regA);
    // qkv = qkv_w @ xln + b  -> (B,P,384) bf16 (B)
    gemm_wmma<0><<<dim3(3, 1024, NB), blk, 0, stream>>>(
        qkvWb, regA, qkv_b, regB, nullptr, nullptr, 128, 384);
    // local attention -> (B,P,128) bf16 (A, reuse)
    attn_kernel<<<8192, blk, 0, stream>>>(regB, regA);
    // y = x + proj_w @ attn + b -> d_out (B,C,HW) f32
    gemm_wmma<2><<<dim3(1, 1024, NB), blk, 0, stream>>>(
        projWb, regA, proj_b, nullptr, x, out, 128, 128);
    // y -> LN2 -> xln2 (A, reuse)
    ln_kernel<<<2048, blk, 0, stream>>>(out, n2_w, n2_b, regA);
    // h = gelu(mlp_w1 @ xln2 + b1) -> (B,P,512) bf16 (B, reuse)
    gemm_wmma<1><<<dim3(4, 1024, NB), blk, 0, stream>>>(
        mlp1Wb, regA, mlp_b1, regB, nullptr, nullptr, 128, 512);
    // out = y + mlp_w2 @ h + b2 (in-place accumulate into d_out)
    gemm_wmma<2><<<dim3(1, 1024, NB), blk, 0, stream>>>(
        mlp2Wb, regB, mlp_b2, nullptr, out, out, 512, 128);
}
